// DeepTTG_37391985279299
// MI455X (gfx1250) — matrix operand order repaired and checked by reading, not compiled, not run on hardware
//
#include <hip/hip_runtime.h>
#include <hip/hip_bf16.h>
#include <math.h>

// ---------------------------------------------------------------------------
// MI455X (gfx1250) implementation.
//
// Live subgraph only: the reference's final attention has KV sequence length 1
// (kv2 = graph_feat[:,None,:]), so softmax==1 and the output depends solely on
// the GIN+GCN graph branch. Everything else (transformer encoders, convs,
// cross-attn, proj) is dead code w.r.t. d_out and is eliminated.
//
// GEMMs run as bf16 WMMA (v_wmma_f32_16x16x32_bf16, f32 accum). Tiles are
// staged in LDS *pre-shuffled into WMMA fragment order* so each lane reads its
// 16-element fragment with two ds_load_b128; global->LDS goes through
// global_load_b128 + packed bf16 ds_store_b64.
//
// Edge aggregation uses f32 global atomics: node feature maps (<=86MB) fit in
// the 192MB L2, whose atomic units service the ~480M scattered adds.
// ---------------------------------------------------------------------------

typedef __attribute__((ext_vector_type(16))) __bf16 v16bf;
typedef __attribute__((ext_vector_type(8)))  __bf16 v8bf;
typedef __attribute__((ext_vector_type(4)))  __bf16 v4bf;
typedef __attribute__((ext_vector_type(8)))  float  v8f;
typedef __attribute__((ext_vector_type(4)))  float  v4f;

#define NN 200000      // nodes
#define NE 800000      // edges
#define BB 128         // batch (graphs)
#define CF 108         // input node features
#define MLP 96         // GIN hidden
#define GCNH 64        // GCN hidden
#define DM 120         // model dim

// ------------------------------ elementwise --------------------------------

__global__ void k_zero(float* __restrict__ p, long long n) {
  long long i = (long long)blockIdx.x * blockDim.x + threadIdx.x;
  if (i < n) p[i] = 0.f;
}

__global__ void k_add(const float* __restrict__ a, const float* __restrict__ b,
                      float* __restrict__ o, long long n) {
  long long i = (long long)blockIdx.x * blockDim.x + threadIdx.x;
  if (i < n) o[i] = a[i] + b[i];
}

// h = (h - m) * rsqrt(v + 1e-5) * g + b   (eval-mode batchnorm, after relu)
__global__ void k_bn(float* __restrict__ h, const float* __restrict__ g,
                     const float* __restrict__ b, const float* __restrict__ m,
                     const float* __restrict__ v, long long n, int F) {
  long long i = (long long)blockIdx.x * blockDim.x + threadIdx.x;
  if (i >= n) return;
  int f = (int)(i % F);
  float x = h[i];
  h[i] = (x - m[f]) * rsqrtf(v[f] + 1e-5f) * g[f] + b[f];
}

// gfeat = alpha*gin + beta*gcn  (alpha/beta are 1-element device arrays)
__global__ void k_combine_gf(const float* __restrict__ gin, const float* __restrict__ gcn,
                             const float* __restrict__ alpha, const float* __restrict__ beta,
                             float* __restrict__ gf, int n) {
  int i = blockIdx.x * blockDim.x + threadIdx.x;
  if (i < n) gf[i] = alpha[0] * gin[i] + beta[0] * gcn[i];
}

// ------------------------------ scatter ops --------------------------------

// out[dst[e], f] += feat[src[e], f]
__global__ void k_edge_gather_add(const float* __restrict__ feat, const int* __restrict__ src,
                                  const int* __restrict__ dst, float* __restrict__ out,
                                  long long nE, int F) {
  long long i = (long long)blockIdx.x * blockDim.x + threadIdx.x;
  if (i >= nE * F) return;
  long long e = i / F;
  int f = (int)(i - e * F);
  atomicAdd(&out[(long long)dst[e] * F + f], feat[(long long)src[e] * F + f]);
}

__global__ void k_deg(const int* __restrict__ dst, float* __restrict__ deg, long long nE) {
  long long e = (long long)blockIdx.x * blockDim.x + threadIdx.x;
  if (e < nE) atomicAdd(&deg[dst[e]], 1.f);
}

__global__ void k_dinv(const float* __restrict__ deg, float* __restrict__ dinv, long long n) {
  long long i = (long long)blockIdx.x * blockDim.x + threadIdx.x;
  if (i < n) dinv[i] = rsqrtf(deg[i] + 1.f);
}

// out[dst, f] += dinv[src]*dinv[dst]*hw[src, f]
__global__ void k_gcn_msg(const float* __restrict__ hw, const int* __restrict__ src,
                          const int* __restrict__ dst, const float* __restrict__ dinv,
                          float* __restrict__ out, long long nE, int F) {
  long long i = (long long)blockIdx.x * blockDim.x + threadIdx.x;
  if (i >= nE * F) return;
  long long e = i / F;
  int f = (int)(i - e * F);
  int s = src[e], d = dst[e];
  atomicAdd(&out[(long long)d * F + f], dinv[s] * dinv[d] * hw[(long long)s * F + f]);
}

// g = relu(agg + hw*dinv^2 + bias)
__global__ void k_gcn_combine(const float* __restrict__ agg, const float* __restrict__ hw,
                              const float* __restrict__ dinv, const float* __restrict__ bias,
                              float* __restrict__ out, long long n, int F) {
  long long i = (long long)blockIdx.x * blockDim.x + threadIdx.x;
  if (i >= n) return;
  int f = (int)(i % F);
  long long node = i / F;
  float di = dinv[node];
  float v = agg[i] + hw[i] * di * di + bias[f];
  out[i] = fmaxf(v, 0.f);
}

// pool[batch[node], f] += h[node, f]
__global__ void k_batch_pool(const float* __restrict__ h, const int* __restrict__ batch,
                             float* __restrict__ pool, long long n, int F) {
  long long i = (long long)blockIdx.x * blockDim.x + threadIdx.x;
  if (i >= n) return;
  long long node = i / F;
  int f = (int)(i - node * F);
  atomicAdd(&pool[(long long)batch[node] * F + f], h[i]);
}

// ------------------------------ WMMA GEMM ----------------------------------
// C[M,N] = act( A[M,K] @ W[N,K]^T + bias )
// Block: 256 threads = 8 wave32s -> 64x64 output tile.
// Waves arranged 4(M) x 2(N); each wave owns a 16x32 strip (two 16x16 WMMAs).
//
// LDS holds tiles in *fragment order*: AsF[strip][lane][p] so that a lane's
// v16bf WMMA operand is 16 contiguous bf16 (two ds_load_b128; lane stride
// padded to 48B for 16B alignment). Cooperative loads use float4
// (global_load_b128) and packed 4xbf16 ds_store_b64 on the full-tile fast
// path. All lda/K values used are multiples of 4 so float4 loads are aligned.
//
// Fragment-position mapping (inverse of ISA 7.12.2 layouts):
//  A tile elem (r,c): strip=r>>4, lane=(r&15)|(((c>>3)&1)<<4),
//                     p = 2*(((c>>4)<<2)|((c&7)>>1)) + (c&1)
//  B tile elem (n,k): tile=n>>4,  lane=(n&15)|(((k>>3)&1)<<4), p = ((k>>4)<<3)|(k&7)

__device__ inline v4bf pk4(v4f x) {
  v4bf r;
  r[0] = (__bf16)x[0]; r[1] = (__bf16)x[1];
  r[2] = (__bf16)x[2]; r[3] = (__bf16)x[3];
  return r;
}

union FragU { v16bf v; v8bf h[2]; };

#define FRAG_STRIDE 24  // __bf16 elements per lane slot (48B, 16B-aligned)

__global__ __launch_bounds__(256) void k_gemm_bf16(
    const float* __restrict__ A, const float* __restrict__ W,
    const float* __restrict__ bias, float* __restrict__ C,
    int M, int N, int K, int lda, int act) {
  __shared__ alignas(16) __bf16 AsF[4][32][FRAG_STRIDE];
  __shared__ alignas(16) __bf16 BsF[4][32][FRAG_STRIDE];

  const int tid  = threadIdx.x;
  const int wave = tid >> 5;
  const int lane = tid & 31;
  const int wm   = wave & 3;   // 0..3 -> M sub-tile (16 rows)
  const int wn   = wave >> 2;  // 0..1 -> N sub-tile (32 cols)

  const int rowBase = blockIdx.y * 64;
  const int colBase = blockIdx.x * 64;

  v8f acc0 = {0.f, 0.f, 0.f, 0.f, 0.f, 0.f, 0.f, 0.f};
  v8f acc1 = {0.f, 0.f, 0.f, 0.f, 0.f, 0.f, 0.f, 0.f};

  for (int k0 = 0; k0 < K; k0 += 32) {
    const bool kFull = (k0 + 32 <= K);

    // ---- A tile (64 rows x 32 k) -> fragment-order LDS ----
    if (kFull && (rowBase + 64 <= M)) {
#pragma unroll
      for (int it = 0; it < 2; ++it) {
        int idx = tid + it * 256;          // 0..511 float4s
        int r = idx >> 3;                  // 0..63
        int c = (idx & 7) << 2;            // 0,4,...,28
        v4f d = *(const v4f*)(A + (long long)(rowBase + r) * lda + (k0 + c));
        int half = (c >> 3) & 1, rem = c & 7;
        int p0 = (((c >> 4) << 2) | (rem >> 1)) << 1;
        *(v4bf*)&AsF[r >> 4][(r & 15) | (half << 4)][p0] = pk4(d);
      }
    } else {
#pragma unroll
      for (int it = 0; it < 8; ++it) {
        int idx = tid + it * 256;          // 0..2047
        int r = idx >> 5, c = idx & 31;
        int gr = rowBase + r, gk = k0 + c;
        float v = (gr < M && gk < K) ? A[(long long)gr * lda + gk] : 0.f;
        int half = (c >> 3) & 1, rem = c & 7;
        int p = ((((c >> 4) << 2) | (rem >> 1)) << 1) | (rem & 1);
        AsF[r >> 4][(r & 15) | (half << 4)][p] = (__bf16)v;
      }
    }

    // ---- B tile (64 cols x 32 k) from W[N,K] -> fragment-order LDS ----
    if (kFull && (colBase + 64 <= N)) {
#pragma unroll
      for (int it = 0; it < 2; ++it) {
        int idx = tid + it * 256;          // 0..511 float4s
        int n = idx >> 3;                  // 0..63
        int kq = (idx & 7) << 2;           // 0,4,...,28
        v4f d = *(const v4f*)(W + (long long)(colBase + n) * K + (k0 + kq));
        *(v4bf*)&BsF[n >> 4][(n & 15) | (((kq >> 3) & 1) << 4)][((kq >> 4) << 3) | (kq & 7)] = pk4(d);
      }
    } else {
#pragma unroll
      for (int it = 0; it < 8; ++it) {
        int idx = tid + it * 256;          // 0..2047
        int n = idx >> 5, kk = idx & 31;
        int gn = colBase + n, gk = k0 + kk;
        float v = (gn < N && gk < K) ? W[(long long)gn * K + gk] : 0.f;
        BsF[n >> 4][(n & 15) | (((kk >> 3) & 1) << 4)][((kk >> 4) << 3) | (kk & 7)] = (__bf16)v;
      }
    }
    __syncthreads();

    // ---- fragments: 16 contiguous bf16 per lane (2x ds_load_b128 each) ----
    FragU fa, fb0, fb1;
    fa.h[0]  = *(const v8bf*)&AsF[wm][lane][0];
    fa.h[1]  = *(const v8bf*)&AsF[wm][lane][8];
    fb0.h[0] = *(const v8bf*)&BsF[wn * 2][lane][0];
    fb0.h[1] = *(const v8bf*)&BsF[wn * 2][lane][8];
    fb1.h[0] = *(const v8bf*)&BsF[wn * 2 + 1][lane][0];
    fb1.h[1] = *(const v8bf*)&BsF[wn * 2 + 1][lane][8];

    acc0 = __builtin_amdgcn_wmma_f32_16x16x32_bf16(false, fa.v, false, fb0.v,
                                                   (short)0, acc0, false, false);
    acc1 = __builtin_amdgcn_wmma_f32_16x16x32_bf16(false, fa.v, false, fb1.v,
                                                   (short)0, acc1, false, false);
    __syncthreads();
  }

  // ---- epilogue: C layout VGPR vi -> row = vi + (lane>>4)*8; col = lane&15
  const int row0 = rowBase + wm * 16 + (lane >> 4) * 8;
  const int c0 = colBase + wn * 32 + (lane & 15);
  const int c1 = c0 + 16;
#pragma unroll
  for (int vi = 0; vi < 8; ++vi) {
    int r = row0 + vi;
    if (r < M) {
      if (c0 < N) {
        float val = acc0[vi];
        if (bias) val += bias[c0];
        if (act) val = fmaxf(val, 0.f);
        C[(long long)r * N + c0] = val;
      }
      if (c1 < N) {
        float val = acc1[vi];
        if (bias) val += bias[c1];
        if (act) val = fmaxf(val, 0.f);
        C[(long long)r * N + c1] = val;
      }
    }
  }
}

// ------------------------------ host driver --------------------------------

extern "C" void kernel_launch(void* const* d_in, const int* in_sizes, int n_in,
                              void* d_out, int out_size, void* d_ws, size_t ws_size,
                              hipStream_t stream) {
  (void)in_sizes; (void)n_in; (void)out_size; (void)ws_size;

  const float* x   = (const float*)d_in[0];
  const int* ei    = (const int*)d_in[1];
  const int* batch = (const int*)d_in[2];
  const int* src = ei;        // edge_index[0]
  const int* dst = ei + NE;   // edge_index[1]

  auto pf = [&](int i) { return (const float*)d_in[i]; };
  // params dict order: 5..27 are embedding/encoder/conv/ckv (dead), then:
  const float* gin_w1[4]; const float* gin_b1[4];
  const float* gin_w2[4]; const float* gin_b2[4];
  const float* bn_g[4];   const float* bn_b[4];
  const float* bn_m[4];   const float* bn_v[4];
  for (int i = 0; i < 4; ++i) {
    int b0 = 28 + i * 8;
    gin_w1[i] = pf(b0 + 0); gin_b1[i] = pf(b0 + 1);
    gin_w2[i] = pf(b0 + 2); gin_b2[i] = pf(b0 + 3);
    bn_g[i] = pf(b0 + 4); bn_b[i] = pf(b0 + 5);
    bn_m[i] = pf(b0 + 6); bn_v[i] = pf(b0 + 7);
  }
  const float* fc1c_w = pf(60); const float* fc1c_b = pf(61);
  const float* gcn1_w = pf(62); const float* gcn1_b = pf(63);
  const float* gcn2_w = pf(64); const float* gcn2_b = pf(65);
  const float* fcgcn_w = pf(66); const float* fcgcn_b = pf(67);
  const float* alpha = pf(70);  const float* beta = pf(71);
  const float* fca_in_w = pf(74); const float* fca_in_b = pf(75);
  const float* fca_out_w = pf(76); const float* fca_out_b = pf(77);
  const float* fc1_w = pf(78); const float* fc1_b = pf(79);
  const float* fc2_w = pf(80); const float* fc2_b = pf(81);
  const float* outw = pf(82);  const float* outb = pf(83);

  // bump allocator in d_ws
  char* wsb = (char*)d_ws;
  size_t off = 0;
  auto alloc = [&](size_t bytes) -> float* {
    float* p = (float*)(wsb + off);
    off += (bytes + 255) & ~(size_t)255;
    return p;
  };
  float* ginpool  = alloc((size_t)BB * MLP * 4);
  float* gcnpool  = alloc((size_t)BB * GCNH * 4);
  float* gin_feat = alloc((size_t)BB * DM * 4);
  float* gcn_feat = alloc((size_t)BB * DM * 4);
  float* gfeat    = alloc((size_t)BB * DM * 4);
  float* vf       = alloc((size_t)BB * DM * 4);
  float* att      = alloc((size_t)BB * DM * 4);
  float* h64      = alloc((size_t)BB * 64 * 4);
  float* h32      = alloc((size_t)BB * 32 * 4);
  float* agg  = alloc((size_t)NN * CF * 4);   // reused as GCN hw
  float* hin  = alloc((size_t)NN * CF * 4);   // reused as GCN agg
  float* tbuf = alloc((size_t)NN * MLP * 4);  // reused as GCN g
  float* hbuf = alloc((size_t)NN * MLP * 4);
  float* deg  = alloc((size_t)NN * 4);
  float* dinv = alloc((size_t)NN * 4);

  auto g1 = [](long long n) { return dim3((unsigned)((n + 255) / 256)); };
  auto zero = [&](float* p, long long n) {
    k_zero<<<g1(n), 256, 0, stream>>>(p, n);
  };
  auto gemm = [&](const float* A, const float* W, const float* bias, float* C,
                  int M, int N, int K, int lda, int act) {
    dim3 grid((unsigned)((N + 63) / 64), (unsigned)((M + 63) / 64), 1);
    k_gemm_bf16<<<grid, 256, 0, stream>>>(A, W, bias, C, M, N, K, lda, act);
  };

  // ---------------- GIN: 4 layers ----------------
  const float* h = x;
  int F = CF;
  for (int i = 0; i < 4; ++i) {
    long long nF = (long long)NN * F;
    zero(agg, nF);
    k_edge_gather_add<<<g1((long long)NE * F), 256, 0, stream>>>(h, src, dst, agg, NE, F);
    k_add<<<g1(nF), 256, 0, stream>>>(h, agg, hin, nF);
    gemm(hin, gin_w1[i], gin_b1[i], tbuf, NN, MLP, F, F, 1);      // relu
    gemm(tbuf, gin_w2[i], gin_b2[i], hbuf, NN, MLP, MLP, MLP, 1); // relu
    k_bn<<<g1((long long)NN * MLP), 256, 0, stream>>>(hbuf, bn_g[i], bn_b[i],
                                                      bn_m[i], bn_v[i],
                                                      (long long)NN * MLP, MLP);
    h = hbuf;
    F = MLP;
  }
  zero(ginpool, (long long)BB * MLP);
  k_batch_pool<<<g1((long long)NN * MLP), 256, 0, stream>>>(hbuf, batch, ginpool,
                                                            (long long)NN * MLP, MLP);
  gemm(ginpool, fc1c_w, fc1c_b, gin_feat, BB, DM, MLP, MLP, 1);  // relu

  // ---------------- GCN: 2 layers (reuse GIN scratch) ----------------
  float* hw   = agg;   // N x 64
  float* gagg = hin;   // N x 64
  float* gbuf = tbuf;  // N x 64

  zero(deg, NN);
  k_deg<<<g1(NE), 256, 0, stream>>>(dst, deg, NE);
  k_dinv<<<g1(NN), 256, 0, stream>>>(deg, dinv, NN);

  gemm(x, gcn1_w, nullptr, hw, NN, GCNH, CF, CF, 0);
  zero(gagg, (long long)NN * GCNH);
  k_gcn_msg<<<g1((long long)NE * GCNH), 256, 0, stream>>>(hw, src, dst, dinv, gagg, NE, GCNH);
  k_gcn_combine<<<g1((long long)NN * GCNH), 256, 0, stream>>>(gagg, hw, dinv, gcn1_b, gbuf,
                                                              (long long)NN * GCNH, GCNH);
  gemm(gbuf, gcn2_w, nullptr, hw, NN, GCNH, GCNH, GCNH, 0);
  zero(gagg, (long long)NN * GCNH);
  k_gcn_msg<<<g1((long long)NE * GCNH), 256, 0, stream>>>(hw, src, dst, dinv, gagg, NE, GCNH);
  k_gcn_combine<<<g1((long long)NN * GCNH), 256, 0, stream>>>(gagg, hw, dinv, gcn2_b, gbuf,
                                                              (long long)NN * GCNH, GCNH);
  zero(gcnpool, (long long)BB * GCNH);
  k_batch_pool<<<g1((long long)NN * GCNH), 256, 0, stream>>>(gbuf, batch, gcnpool,
                                                             (long long)NN * GCNH, GCNH);
  gemm(gcnpool, fcgcn_w, fcgcn_b, gcn_feat, BB, DM, GCNH, GCNH, 1);  // relu

  // ---------------- graph_feat = alpha*gin + beta*gcn ----------------
  k_combine_gf<<<g1(BB * DM), 256, 0, stream>>>(gin_feat, gcn_feat, alpha, beta,
                                                gfeat, BB * DM);

  // ---------------- head ----------------
  // Final MHA has KV len 1 => softmax==1 => output = out_proj(v_proj(graph_feat))
  gemm(gfeat, fca_in_w + 240 * DM, fca_in_b + 240, vf, BB, DM, DM, DM, 0);
  gemm(vf, fca_out_w, fca_out_b, att, BB, DM, DM, DM, 0);
  gemm(att, fc1_w, fc1_b, h64, BB, 64, DM, DM, 1);
  gemm(h64, fc2_w, fc2_b, h32, BB, 32, 64, 64, 1);
  gemm(h32, outw, outb, (float*)d_out, BB, 1, 32, 32, 0);
}
